// DeformableTransformer_62586263437927
// MI455X (gfx1250) — matrix operand order repaired and checked by reading, not compiled, not run on hardware
//
#include <hip/hip_runtime.h>
#include <math.h>

#define NBATCH 4
#define QLEN   300
#define CDIM   256
#define NHEAD  8
#define DHEAD  32
#define NLEV   4
#define NPTS   4
#define DFFN   1024
#define NLAYER 6
#define STOT   13294
#define NQTOT  (NBATCH*QLEN)
#define TOPKN  30

typedef __attribute__((ext_vector_type(16))) __bf16 v16bf;
typedef __attribute__((ext_vector_type(8)))  float  v8f;

__constant__ int c_HS[4] = {100, 50, 25, 13};
__constant__ int c_WS[4] = {100, 50, 25, 13};
__constant__ int c_ST[4] = {0, 10000, 12500, 13125};

__device__ __forceinline__ unsigned short f2bf(float f) {
  unsigned u = __float_as_uint(f);
  u += 0x7fffu + ((u >> 16) & 1u);   // round-to-nearest-even
  return (unsigned short)(u >> 16);
}

union BF16Frag {
  v16bf v;
  unsigned short u[16];
  uint4 q[2];
};

// ---------------------------------------------------------------------------
// Pack fp32 A[M,K] into WMMA A-fragment layout (bf16), tile = 16Mx32K.
// ISA 16-bit A layout: lane holds row (lane&15); half-wave (lane>>4) selects
// K offset {0..7} / {8..15}; elements 0..7 <-> K = kb+e, 8..15 <-> K = kb+16+e.
// Flat: [mt][kt][lane][16 halfs].
// ---------------------------------------------------------------------------
__global__ void pack_a_bf16(const float* __restrict__ A, unsigned short* __restrict__ P,
                            int M, int K)
{
  const int tiles_k = K >> 5;
  const int mtiles  = (M + 15) >> 4;
  const int total   = mtiles * tiles_k * 32;
  int gid = blockIdx.x * blockDim.x + threadIdx.x;
  if (gid >= total) return;
  const int lane = gid & 31;
  const int ti = gid >> 5;
  const int kt = ti % tiles_k;
  const int mt = ti / tiles_k;
  int row = mt * 16 + (lane & 15);
  if (row >= M) row = M - 1;                       // edge rows duplicated; stores guarded
  const int kb = kt * 32 + (lane >> 4) * 8;
  const float* srcp = A + (size_t)row * K;
  unsigned short* dst = P + (size_t)gid * 16;
  #pragma unroll
  for (int e = 0; e < 8; ++e) dst[e] = f2bf(srcp[kb + e]);
  #pragma unroll
  for (int e = 0; e < 8; ++e) dst[8 + e] = f2bf(srcp[kb + 16 + e]);
}

// ---------------------------------------------------------------------------
// Pack fp32 weights into WMMA B-fragment layout (bf16), tile = 32Kx16N.
// lane = column (lane&15); half-wave selects K offset 0/8; elements 0..7 <-> K+e, 8..15 <-> K+16+(e-8).
// Flat: [tn][tk][lane][16 halfs].
// ---------------------------------------------------------------------------
__global__ void pack_b_bf16(const float* __restrict__ B, unsigned short* __restrict__ P,
                            int N, int K, int ldb, int transB)
{
  const int tiles_k = K >> 5;
  const int total = (N >> 4) * tiles_k * 32;
  int gid = blockIdx.x * blockDim.x + threadIdx.x;
  if (gid >= total) return;
  const int lane = gid & 31;
  const int ti = gid >> 5;
  const int tk = ti % tiles_k;
  const int tn = ti / tiles_k;
  const int col = tn * 16 + (lane & 15);
  const int kb  = tk * 32 + (lane >> 4) * 8;
  unsigned short* dst = P + (size_t)gid * 16;
  #pragma unroll
  for (int e = 0; e < 16; ++e) {
    const int k = kb + (e < 8 ? e : e + 8);
    const float f = transB ? B[(size_t)col * ldb + k] : B[(size_t)k * ldb + col];
    dst[e] = f2bf(f);
  }
}

// ---------------------------------------------------------------------------
// GEMM: C = A(packed bf16) * B(packed bf16) + bias, opt ReLU, f32 accumulate.
// Wave computes a 32(M) x 128(N) strip: 2 A frags x 8 B frags -> 16 WMMA/K-step
// (20 b128 loads per 16 WMMA). Named B pointers keep loads in global space.
// ---------------------------------------------------------------------------
__global__ __launch_bounds__(256) void gemm_wmma_bf16(
    const unsigned short* __restrict__ Ap, const unsigned short* __restrict__ Bp,
    const float* __restrict__ bias, float* __restrict__ C,
    int M, int K, int relu, int ldc)
{
  const int lane = threadIdx.x & 31;
  const int wave = threadIdx.x >> 5;
  const int m0 = (blockIdx.y * 8 + wave) * 32;
  const int n0 = blockIdx.x * 128;
  if (m0 >= M) return;
  const int hi = lane >> 4;
  const int tiles_k = K >> 5;
  const int mtiles = (M + 15) >> 4;
  const int mt0 = m0 >> 4;
  int mt1 = mt0 + 1;
  if (mt1 > mtiles - 1) mt1 = mtiles - 1;          // clamp; EXEC stays all-ones

  v8f acc[2][8];
  #pragma unroll
  for (int s = 0; s < 2; ++s)
    #pragma unroll
    for (int t = 0; t < 8; ++t)
      #pragma unroll
      for (int r = 0; r < 8; ++r) acc[s][t][r] = 0.0f;

  const unsigned short* a0p = Ap + ((size_t)mt0 * tiles_k * 32 + lane) * 16;
  const unsigned short* a1p = Ap + ((size_t)mt1 * tiles_k * 32 + lane) * 16;
  const unsigned short* bp0 = Bp + (((size_t)((n0 >> 4) + 0) * tiles_k) * 32 + lane) * 16;
  const unsigned short* bp1 = Bp + (((size_t)((n0 >> 4) + 1) * tiles_k) * 32 + lane) * 16;
  const unsigned short* bp2 = Bp + (((size_t)((n0 >> 4) + 2) * tiles_k) * 32 + lane) * 16;
  const unsigned short* bp3 = Bp + (((size_t)((n0 >> 4) + 3) * tiles_k) * 32 + lane) * 16;
  const unsigned short* bp4 = Bp + (((size_t)((n0 >> 4) + 4) * tiles_k) * 32 + lane) * 16;
  const unsigned short* bp5 = Bp + (((size_t)((n0 >> 4) + 5) * tiles_k) * 32 + lane) * 16;
  const unsigned short* bp6 = Bp + (((size_t)((n0 >> 4) + 6) * tiles_k) * 32 + lane) * 16;
  const unsigned short* bp7 = Bp + (((size_t)((n0 >> 4) + 7) * tiles_k) * 32 + lane) * 16;

  for (int kt = 0; kt < tiles_k; ++kt) {
    __builtin_prefetch(a0p + 512, 0, 1);           // global_prefetch for streamed A
    __builtin_prefetch(a1p + 512, 0, 1);
    BF16Frag a0, a1, b;
    a0.q[0] = *(const uint4*)(a0p);
    a0.q[1] = *(const uint4*)(a0p + 8);
    a1.q[0] = *(const uint4*)(a1p);
    a1.q[1] = *(const uint4*)(a1p + 8);
    a0p += 512; a1p += 512;

    b.q[0] = *(const uint4*)(bp0); b.q[1] = *(const uint4*)(bp0 + 8); bp0 += 512;
    acc[0][0] = __builtin_amdgcn_wmma_f32_16x16x32_bf16(false, a0.v, false, b.v, (short)0, acc[0][0], false, false);
    acc[1][0] = __builtin_amdgcn_wmma_f32_16x16x32_bf16(false, a1.v, false, b.v, (short)0, acc[1][0], false, false);

    b.q[0] = *(const uint4*)(bp1); b.q[1] = *(const uint4*)(bp1 + 8); bp1 += 512;
    acc[0][1] = __builtin_amdgcn_wmma_f32_16x16x32_bf16(false, a0.v, false, b.v, (short)0, acc[0][1], false, false);
    acc[1][1] = __builtin_amdgcn_wmma_f32_16x16x32_bf16(false, a1.v, false, b.v, (short)0, acc[1][1], false, false);

    b.q[0] = *(const uint4*)(bp2); b.q[1] = *(const uint4*)(bp2 + 8); bp2 += 512;
    acc[0][2] = __builtin_amdgcn_wmma_f32_16x16x32_bf16(false, a0.v, false, b.v, (short)0, acc[0][2], false, false);
    acc[1][2] = __builtin_amdgcn_wmma_f32_16x16x32_bf16(false, a1.v, false, b.v, (short)0, acc[1][2], false, false);

    b.q[0] = *(const uint4*)(bp3); b.q[1] = *(const uint4*)(bp3 + 8); bp3 += 512;
    acc[0][3] = __builtin_amdgcn_wmma_f32_16x16x32_bf16(false, a0.v, false, b.v, (short)0, acc[0][3], false, false);
    acc[1][3] = __builtin_amdgcn_wmma_f32_16x16x32_bf16(false, a1.v, false, b.v, (short)0, acc[1][3], false, false);

    b.q[0] = *(const uint4*)(bp4); b.q[1] = *(const uint4*)(bp4 + 8); bp4 += 512;
    acc[0][4] = __builtin_amdgcn_wmma_f32_16x16x32_bf16(false, a0.v, false, b.v, (short)0, acc[0][4], false, false);
    acc[1][4] = __builtin_amdgcn_wmma_f32_16x16x32_bf16(false, a1.v, false, b.v, (short)0, acc[1][4], false, false);

    b.q[0] = *(const uint4*)(bp5); b.q[1] = *(const uint4*)(bp5 + 8); bp5 += 512;
    acc[0][5] = __builtin_amdgcn_wmma_f32_16x16x32_bf16(false, a0.v, false, b.v, (short)0, acc[0][5], false, false);
    acc[1][5] = __builtin_amdgcn_wmma_f32_16x16x32_bf16(false, a1.v, false, b.v, (short)0, acc[1][5], false, false);

    b.q[0] = *(const uint4*)(bp6); b.q[1] = *(const uint4*)(bp6 + 8); bp6 += 512;
    acc[0][6] = __builtin_amdgcn_wmma_f32_16x16x32_bf16(false, a0.v, false, b.v, (short)0, acc[0][6], false, false);
    acc[1][6] = __builtin_amdgcn_wmma_f32_16x16x32_bf16(false, a1.v, false, b.v, (short)0, acc[1][6], false, false);

    b.q[0] = *(const uint4*)(bp7); b.q[1] = *(const uint4*)(bp7 + 8); bp7 += 512;
    acc[0][7] = __builtin_amdgcn_wmma_f32_16x16x32_bf16(false, a0.v, false, b.v, (short)0, acc[0][7], false, false);
    acc[1][7] = __builtin_amdgcn_wmma_f32_16x16x32_bf16(false, a1.v, false, b.v, (short)0, acc[1][7], false, false);
  }

  // D layout: lane (l&15) = column, VGPR r -> row base + r + 8*(lane>>4)
  #pragma unroll
  for (int s = 0; s < 2; ++s) {
    #pragma unroll
    for (int t = 0; t < 8; ++t) {
      const int col = n0 + t * 16 + (lane & 15);
      const float bb = bias ? bias[col] : 0.0f;
      #pragma unroll
      for (int r = 0; r < 8; ++r) {
        const int m = m0 + s * 16 + hi * 8 + r;
        if (m < M) {
          float vv = acc[s][t][r] + bb;
          if (relu) vv = fmaxf(vv, 0.0f);
          C[(size_t)m * ldc + col] = vv;
        }
      }
    }
  }
}

// ---------------------------------------------------------------------------
// Elementwise helpers
// ---------------------------------------------------------------------------
__global__ void copy_kern(const float* __restrict__ a, float* __restrict__ b, int n) {
  int i = blockIdx.x * blockDim.x + threadIdx.x;
  if (i < n) b[i] = a[i];
}
__global__ void add_kern(const float* __restrict__ a, const float* __restrict__ b,
                         float* __restrict__ c, int n) {
  int i = blockIdx.x * blockDim.x + threadIdx.x;
  if (i < n) c[i] = a[i] + b[i];
}

// Fused residual + LayerNorm over CDIM=256 (one block per row, 8 waves)
__global__ __launch_bounds__(256) void res_ln(
    const float* __restrict__ x, const float* __restrict__ r,
    const float* __restrict__ g, const float* __restrict__ b, float* __restrict__ dst)
{
  const int row = blockIdx.x, t = threadIdx.x;
  __shared__ float red[8];
  float v = x[(size_t)row * CDIM + t] + r[(size_t)row * CDIM + t];
  float s = v;
  #pragma unroll
  for (int m = 16; m > 0; m >>= 1) s += __shfl_xor(s, m);
  if ((t & 31) == 0) red[t >> 5] = s;
  __syncthreads();
  float mean = 0.f;
  #pragma unroll
  for (int i = 0; i < 8; ++i) mean += red[i];
  mean *= (1.0f / CDIM);
  __syncthreads();
  const float dv = v - mean;
  float s2 = dv * dv;
  #pragma unroll
  for (int m = 16; m > 0; m >>= 1) s2 += __shfl_xor(s2, m);
  if ((t & 31) == 0) red[t >> 5] = s2;
  __syncthreads();
  float var = 0.f;
  #pragma unroll
  for (int i = 0; i < 8; ++i) var += red[i];
  var *= (1.0f / CDIM);
  dst[(size_t)row * CDIM + t] = dv * rsqrtf(var + 1e-5f) * g[t] + b[t];
}

// ---------------------------------------------------------------------------
// Self-attention: one block per (q, h, n); qkv = [NQ, 768] (Q|K|V concat)
// ---------------------------------------------------------------------------
__global__ __launch_bounds__(320) void mha_attn(const float* __restrict__ qkv,
                                                float* __restrict__ o)
{
  const int q = blockIdx.x, h = blockIdx.y, n = blockIdx.z;
  const int t = threadIdx.x;
  __shared__ float sq[DHEAD];
  __shared__ float sp[QLEN];
  __shared__ float red[10];
  __shared__ float pbuf[320];
  const float scale = 0.17677669529663687f;  // 32^-0.5
  if (t < DHEAD) sq[t] = qkv[((size_t)(n * QLEN + q)) * 768 + h * DHEAD + t] * scale;
  __syncthreads();
  float sc = -1e30f;
  if (t < QLEN) {
    const float* kr = qkv + ((size_t)(n * QLEN + t)) * 768 + CDIM + h * DHEAD;
    float dsum = 0.f;
    #pragma unroll
    for (int d = 0; d < DHEAD; ++d) dsum += sq[d] * kr[d];
    sc = dsum;
  }
  float v = sc;
  #pragma unroll
  for (int m = 16; m > 0; m >>= 1) v = fmaxf(v, __shfl_xor(v, m));
  if ((t & 31) == 0) red[t >> 5] = v;
  __syncthreads();
  float mx = red[0];
  #pragma unroll
  for (int i = 1; i < 10; ++i) mx = fmaxf(mx, red[i]);
  __syncthreads();
  float p = (t < QLEN) ? __expf(sc - mx) : 0.f;
  if (t < QLEN) sp[t] = p;
  float s = p;
  #pragma unroll
  for (int m = 16; m > 0; m >>= 1) s += __shfl_xor(s, m);
  if ((t & 31) == 0) red[t >> 5] = s;
  __syncthreads();
  float tot = 0.f;
  #pragma unroll
  for (int i = 0; i < 10; ++i) tot += red[i];
  const int d = t & 31, chunk = t >> 5;
  float part = 0.f;
  const int kb = chunk * 30;
  for (int k = 0; k < 30; ++k)
    part += sp[kb + k] * qkv[((size_t)(n * QLEN + kb + k)) * 768 + 2 * CDIM + h * DHEAD + d];
  pbuf[t] = part;
  __syncthreads();
  if (t < DHEAD) {
    float oa = 0.f;
    #pragma unroll
    for (int c = 0; c < 10; ++c) oa += pbuf[c * 32 + t];
    o[((size_t)(n * QLEN + q)) * CDIM + h * DHEAD + t] = oa / tot;
  }
}

// softmax over 16 points per (n,q,h); aw layout [NQ,128]
__global__ void softmax16_kern(float* __restrict__ aw, int n) {
  int i = blockIdx.x * blockDim.x + threadIdx.x;
  if (i >= n) return;
  float* p = aw + (size_t)i * 16;
  float mx = p[0];
  #pragma unroll
  for (int j = 1; j < 16; ++j) mx = fmaxf(mx, p[j]);
  float s = 0.f;
  #pragma unroll
  for (int j = 0; j < 16; ++j) { float e = __expf(p[j] - mx); p[j] = e; s += e; }
  const float inv = 1.0f / s;
  #pragma unroll
  for (int j = 0; j < 16; ++j) p[j] *= inv;
}

// ---------------------------------------------------------------------------
// MSDA sampling: block per (n,q); thread = head (t>>5) x channel (t&31).
// Phase 1 (128 threads): compute sample locations + weights once into LDS
// (and store loc for the top-k output). Phase 2: all 256 threads gather.
// ---------------------------------------------------------------------------
__global__ __launch_bounds__(256) void msda_sample(
    const float* __restrict__ value, const float* __restrict__ off,
    const float* __restrict__ aw, const float* __restrict__ refp,
    const float* __restrict__ vr, float* __restrict__ acc_out,
    float* __restrict__ loc_out)
{
  const int nq = blockIdx.x;
  const int n = nq / QLEN;
  const int t = threadIdx.x;
  const int h = t >> 5, d = t & 31;
  __shared__ float sLx[128], sLy[128], sAw[128];

  if (t < 128) {  // f = t = ((h*4+l)*4+p)
    const int l = (t >> 2) & 3;
    const float Wl = (float)c_WS[l], Hl = (float)c_HS[l];
    const float rx = refp[nq * 2 + 0], ry = refp[nq * 2 + 1];
    const float vrx = vr[(n * NLEV + l) * 2 + 0], vry = vr[(n * NLEV + l) * 2 + 1];
    const float lx = rx * vrx + off[(size_t)nq * 256 + t * 2 + 0] / Wl;
    const float ly = ry * vry + off[(size_t)nq * 256 + t * 2 + 1] / Hl;
    sLx[t] = lx; sLy[t] = ly;
    sAw[t] = aw[(size_t)nq * 128 + t];
    loc_out[((size_t)nq * 128 + t) * 2 + 0] = lx;
    loc_out[((size_t)nq * 128 + t) * 2 + 1] = ly;
  }
  __syncthreads();

  float accv = 0.f;
  for (int l = 0; l < NLEV; ++l) {
    const int Wl = c_WS[l], Hl = c_HS[l], s0 = c_ST[l];
    const float* vbase = value + ((size_t)n * STOT + s0) * CDIM + h * DHEAD + d;
    #pragma unroll
    for (int p = 0; p < NPTS; ++p) {
      const int fi = (h * NLEV + l) * NPTS + p;
      const float gx = sLx[fi] * (float)Wl - 0.5f;
      const float gy = sLy[fi] * (float)Hl - 0.5f;
      const float x0f = floorf(gx), y0f = floorf(gy);
      const float wx = gx - x0f, wy = gy - y0f;
      const int x0 = (int)x0f, y0 = (int)y0f;
      float t00 = 0.f, t10 = 0.f, t01 = 0.f, t11 = 0.f;
      if (x0 >= 0     && x0 < Wl     && y0 >= 0     && y0 < Hl)
        t00 = vbase[(size_t)(y0 * Wl + x0) * CDIM];
      if (x0 + 1 >= 0 && x0 + 1 < Wl && y0 >= 0     && y0 < Hl)
        t10 = vbase[(size_t)(y0 * Wl + x0 + 1) * CDIM];
      if (x0 >= 0     && x0 < Wl     && y0 + 1 >= 0 && y0 + 1 < Hl)
        t01 = vbase[(size_t)((y0 + 1) * Wl + x0) * CDIM];
      if (x0 + 1 >= 0 && x0 + 1 < Wl && y0 + 1 >= 0 && y0 + 1 < Hl)
        t11 = vbase[(size_t)((y0 + 1) * Wl + x0 + 1) * CDIM];
      const float samp = t00 * (1.f - wx) * (1.f - wy) + t10 * wx * (1.f - wy)
                       + t01 * (1.f - wx) * wy + t11 * wx * wy;
      accv += sAw[fi] * samp;
    }
  }
  acc_out[(size_t)nq * CDIM + h * DHEAD + d] = accv;
}

// top-30 of 128 (stable, descending; matches lax.top_k) + valid-ratio divide
__global__ __launch_bounds__(256) void topk_gather(
    const float* __restrict__ aw, const float* __restrict__ loc,
    const float* __restrict__ vr, float* __restrict__ out)
{
  int gid = blockIdx.x * blockDim.x + threadIdx.x;
  if (gid >= NQTOT) return;
  const int n = gid / QLEN;
  float w[128];
  for (int f = 0; f < 128; ++f) w[f] = aw[(size_t)gid * 128 + f];
  unsigned used[4] = {0u, 0u, 0u, 0u};
  for (int s = 0; s < TOPKN; ++s) {
    float best = -3.4e38f; int bi = 0;
    for (int f = 0; f < 128; ++f) {
      if (((used[f >> 5] >> (f & 31)) & 1u) == 0u && w[f] > best) { best = w[f]; bi = f; }
    }
    used[bi >> 5] |= 1u << (bi & 31);
    const int l = (bi >> 2) & 3;
    out[((size_t)gid * TOPKN + s) * 2 + 0] =
        loc[((size_t)gid * 128 + bi) * 2 + 0] / vr[(n * NLEV + l) * 2 + 0];
    out[((size_t)gid * TOPKN + s) * 2 + 1] =
        loc[((size_t)gid * 128 + bi) * 2 + 1] / vr[(n * NLEV + l) * 2 + 1];
  }
}

// ---------------------------------------------------------------------------
extern "C" void kernel_launch(void* const* d_in, const int* in_sizes, int n_in,
                              void* d_out, int out_size, void* d_ws, size_t ws_size,
                              hipStream_t stream)
{
  (void)in_sizes; (void)n_in; (void)out_size; (void)ws_size;
  const float* tgt  = (const float*)d_in[0];
  const float* refp = (const float*)d_in[1];
  const float* src  = (const float*)d_in[2];
  const float* vr   = (const float*)d_in[5];
  const float* qpos = (const float*)d_in[6];
  const float* Wv   = (const float*)d_in[7];
  const float* bv   = (const float*)d_in[8];
  const float* Woff = (const float*)d_in[9];
  const float* boff = (const float*)d_in[10];
  const float* Waw  = (const float*)d_in[11];
  const float* baw  = (const float*)d_in[12];
  const float* Wo   = (const float*)d_in[13];
  const float* bo   = (const float*)d_in[14];
  const float* Wqkv = (const float*)d_in[15];
  const float* bqkv = (const float*)d_in[16];
  const float* Wmo  = (const float*)d_in[17];
  const float* bmo  = (const float*)d_in[18];
  const float* ln1g = (const float*)d_in[19];
  const float* ln1b = (const float*)d_in[20];
  const float* ln2g = (const float*)d_in[21];
  const float* ln2b = (const float*)d_in[22];
  const float* ln3g = (const float*)d_in[23];
  const float* ln3b = (const float*)d_in[24];
  const float* W1   = (const float*)d_in[25];
  const float* b1   = (const float*)d_in[26];
  const float* W2   = (const float*)d_in[27];
  const float* b2   = (const float*)d_in[28];
  float* outp = (float*)d_out;

  float* ws = (float*)d_ws;
  size_t wso = 0;
  auto alloc = [&](size_t nf) { float* p = ws + wso; wso += nf; return p; };
  float* out_cur  = alloc((size_t)NQTOT * CDIM);
  float* q_buf    = alloc((size_t)NQTOT * CDIM);
  float* qkv_buf  = alloc((size_t)NQTOT * 3 * CDIM);
  float* sa_buf   = alloc((size_t)NQTOT * CDIM);
  float* proj_buf = alloc((size_t)NQTOT * CDIM);
  float* acc_buf  = alloc((size_t)NQTOT * CDIM);
  float* off_buf  = alloc((size_t)NQTOT * 256);
  float* aw_buf   = alloc((size_t)NQTOT * 128);
  float* loc_buf  = alloc((size_t)NQTOT * 256);
  float* mid_buf  = alloc((size_t)NQTOT * DFFN);
  float* val_buf  = alloc((size_t)NBATCH * STOT * CDIM);
  // packed-fragment scratch (bf16 stored in float-sized ws)
  const int srcMtiles = (NBATCH * STOT + 15) / 16;                  // 3324
  unsigned short* packSrc = (unsigned short*)alloc((size_t)srcMtiles * 8 * 512 / 2);
  unsigned short* packA   = (unsigned short*)alloc((size_t)76 * 32 * 512 / 2); // up to 1216x1024
  unsigned short* packB   = (unsigned short*)alloc(131072);        // up to 1024x256

  auto packAGo = [&](const float* A, int M, int K, unsigned short* dst) {
    const int total = ((M + 15) / 16) * (K / 32) * 32;
    pack_a_bf16<<<(total + 255) / 256, 256, 0, stream>>>(A, dst, M, K);
  };
  auto gemm = [&](const unsigned short* Ap, const float* Bw, const float* bias, float* Cm,
                  int M, int N, int K, int ldb, int transB, int relu, int ldc) {
    const int totalB = (N / 16) * (K / 32) * 32;
    pack_b_bf16<<<(totalB + 255) / 256, 256, 0, stream>>>(Bw, packB, N, K, ldb, transB);
    dim3 g(N / 128, (M + 255) / 256);
    gemm_wmma_bf16<<<g, 256, 0, stream>>>(Ap, packB, bias, Cm, M, K, relu, ldc);
  };

  const int NE = NQTOT * CDIM;
  copy_kern<<<(NE + 255) / 256, 256, 0, stream>>>(tgt, out_cur, NE);
  packAGo(src, NBATCH * STOT, CDIM, packSrc);      // layer-invariant: pack once

  for (int i = 0; i < NLAYER; ++i) {
    const float* Wqkv_i = Wqkv + (size_t)i * 3 * CDIM * CDIM;
    const float* bqkv_i = bqkv + (size_t)i * 3 * CDIM;

    // ---- self-attention ----
    add_kern<<<(NE + 255) / 256, 256, 0, stream>>>(out_cur, qpos, q_buf, NE);
    packAGo(q_buf, NQTOT, CDIM, packA);
    gemm(packA, Wqkv_i, bqkv_i, qkv_buf, NQTOT, 512, CDIM, CDIM, 1, 0, 768);   // Q,K: x@W.T
    packAGo(out_cur, NQTOT, CDIM, packA);                                      // V from pre-pos out
    gemm(packA, Wqkv_i + 2 * CDIM * CDIM, bqkv_i + 2 * CDIM, qkv_buf + 512,
         NQTOT, 256, CDIM, CDIM, 1, 0, 768);
    mha_attn<<<dim3(QLEN, NHEAD, NBATCH), 320, 0, stream>>>(qkv_buf, sa_buf);
    packAGo(sa_buf, NQTOT, CDIM, packA);
    gemm(packA, Wmo + (size_t)i * CDIM * CDIM, bmo + (size_t)i * CDIM, proj_buf,
         NQTOT, CDIM, CDIM, CDIM, 0, 0, CDIM);
    res_ln<<<NQTOT, CDIM, 0, stream>>>(out_cur, proj_buf, ln2g + i * CDIM, ln2b + i * CDIM, out_cur);

    // ---- MS-deformable cross-attention ----
    gemm(packSrc, Wv + (size_t)i * CDIM * CDIM, bv + (size_t)i * CDIM, val_buf,
         NBATCH * STOT, CDIM, CDIM, CDIM, 0, 0, CDIM);                         // dominant GEMM
    add_kern<<<(NE + 255) / 256, 256, 0, stream>>>(out_cur, qpos, q_buf, NE);
    packAGo(q_buf, NQTOT, CDIM, packA);                                        // shared by off+aw
    gemm(packA, Woff + (size_t)i * CDIM * 256, boff + (size_t)i * 256, off_buf,
         NQTOT, 256, CDIM, 256, 0, 0, 256);
    gemm(packA, Waw + (size_t)i * CDIM * 128, baw + (size_t)i * 128, aw_buf,
         NQTOT, 128, CDIM, 128, 0, 0, 128);
    softmax16_kern<<<(NQTOT * NHEAD + 255) / 256, 256, 0, stream>>>(aw_buf, NQTOT * NHEAD);
    msda_sample<<<NQTOT, 256, 0, stream>>>(val_buf, off_buf, aw_buf, refp, vr, acc_buf, loc_buf);
    packAGo(acc_buf, NQTOT, CDIM, packA);
    gemm(packA, Wo + (size_t)i * CDIM * CDIM, bo + (size_t)i * CDIM, proj_buf,
         NQTOT, CDIM, CDIM, CDIM, 0, 0, CDIM);
    res_ln<<<NQTOT, CDIM, 0, stream>>>(out_cur, proj_buf, ln1g + i * CDIM, ln1b + i * CDIM, out_cur);

    // ---- FFN ----
    packAGo(out_cur, NQTOT, CDIM, packA);
    gemm(packA, W1 + (size_t)i * CDIM * DFFN, b1 + (size_t)i * DFFN, mid_buf,
         NQTOT, DFFN, CDIM, DFFN, 0, 1, DFFN);
    packAGo(mid_buf, NQTOT, DFFN, packA);
    gemm(packA, W2 + (size_t)i * DFFN * CDIM, b2 + (size_t)i * CDIM, proj_buf,
         NQTOT, CDIM, DFFN, CDIM, 0, 0, CDIM);
    res_ln<<<NQTOT, CDIM, 0, stream>>>(out_cur, proj_buf, ln3g + i * CDIM, ln3b + i * CDIM,
                                       (i == NLAYER - 1) ? outp : out_cur);
  }

  // outputs: [out | reference_points | samples_keep]
  copy_kern<<<(NQTOT * 2 + 255) / 256, 256, 0, stream>>>(refp, outp + (size_t)NQTOT * CDIM, NQTOT * 2);
  topk_gather<<<(NQTOT + 255) / 256, 256, 0, stream>>>(
      aw_buf, loc_buf, vr, outp + (size_t)NQTOT * CDIM + NQTOT * 2);
}
